// Convolution_vect_76158360093118
// MI455X (gfx1250) — compile-verified, matrix operand order checked by reading, not run on hardware
//
#include <hip/hip_runtime.h>

// ---------------------------------------------------------------------------
// MACE symmetric contraction:
//   out[n,c] = sum_p f_p * ( UW1[c,p] + sum_q UW2[c,p,q] f_q
//                            + sum_{q,i} UW3[p,q,i,c] f_q f_i )
// => per c: GEMM  D[48 x Ntile] = A_c[48 x 2368] * B[2368 x Ntile]
//    A_c = precomputed weights (bf16 blob, WMMA-fragment layout, ~18 MB -> L2)
//    B   = per-atom outer products f_q*f_i (built in LDS, bf16)
// All matrix math uses v_wmma_f32_16x16x32_bf16 (f32 accumulation).
// ---------------------------------------------------------------------------

#define NA   1024
#define NC   80
#define NI   48
#define PQ   2304          // 48*48
#define K3C  1270
#define KP   2368          // padded K' = 74*32  (2304 + 48 + 1 + 15 pad)
#define NKT  74            // K' tiles of 32
#define FRAG 512           // elements per fragment tile (32 lanes * 16)
#define BLOB_ELEMS ((long)NC * 3 * NKT * FRAG)        // 9,093,120 bf16
#define W3F_ELEMS  (40 * 5 * FRAG)                    // 102,400 bf16

typedef __attribute__((ext_vector_type(16))) __bf16       v16bf;
typedef __attribute__((ext_vector_type(2)))  __bf16       v2bf;
typedef __attribute__((ext_vector_type(8)))  float        v8f;
typedef __attribute__((ext_vector_type(8)))  unsigned int v8u;

static __device__ __forceinline__ unsigned short bfbits(float x) {
  __bf16 h = (__bf16)x;
  return __builtin_bit_cast(unsigned short, h);
}
static __device__ __forceinline__ unsigned pack2(float a, float b) {
  v2bf t; t[0] = (__bf16)a; t[1] = (__bf16)b;     // -> v_cvt_pk_bf16_f32
  return __builtin_bit_cast(unsigned, t);
}
// K position held by fragment element e for half h (ISA 16-bit A 16x32 layout:
// lanes 0-15: K = {0..7, 16..23}; lanes 16-31: +8). B uses the mirrored layout.
static __device__ __forceinline__ int kmap(int e, int h) {
  return (e < 8 ? e : e + 8) + 8 * h;
}
// blob element index for value A_c[p, K]  (inverse of the fragment mapping)
static __device__ __forceinline__ long blob_idx(int c, int p, int K) {
  int mt = p >> 4;
  int kt = K >> 5;
  int kk = K & 31;
  int h  = (kk >> 3) & 1;
  int e  = (kk & 7) + (kk >= 16 ? 8 : 0);
  int lane = h * 16 + (p & 15);
  return ((long)((c * 3 + mt) * NKT + kt) * 32 + lane) * 16 + e;
}

// ---------------------------------------------------------------------------
// Kernel 0: pre-pack W3 into WMMA B-fragments (bf16), shared by all 6912
// row-tile blocks of k_uw3.  w3f[((kt*5+ct)*32+lane)*16+e] = W3[K, ct*16+ln]
// ---------------------------------------------------------------------------
__global__ void k_w3frag(const float* __restrict__ W3,
                         unsigned short* __restrict__ w3f)
{
  int t = blockIdx.x * 256 + threadIdx.x;
  if (t >= W3F_ELEMS) return;
  int e    = t & 15;
  int lane = (t >> 4) & 31;
  int ctkt = t >> 9;
  int ct   = ctkt % 5;
  int kt   = ctkt / 5;
  int h = lane >> 4, ln = lane & 15;
  int K = kt * 32 + kmap(e, h);
  int cc = ct * 16 + ln;
  float v = (K < K3C) ? W3[K * NC + cc] : 0.f;
  w3f[t] = bfbits(v);
}

// ---------------------------------------------------------------------------
// Kernel 1: UW3[p,qi,c] = sum_k U3[pqi,k] * W3[k,c]  via bf16 WMMA.
// One wave per 16-row tile; the A fragment (U3, vector float2 loads, branch-
// free) is reused across five c tiles, so U3 (562 MB) streams from HBM once.
// B fragments come pre-packed from w3f (one 32 B load each, L2-hot).
// ---------------------------------------------------------------------------
__global__ void __launch_bounds__(32)
k_uw3(const float* __restrict__ U3, const unsigned short* __restrict__ w3f,
      unsigned short* __restrict__ blob)
{
  const int lane = threadIdx.x;
  const int rt   = blockIdx.x;            // 0..6911  (110592 rows / 16)
  const int h    = lane >> 4;
  const int ln   = lane & 15;
  const long row = (long)rt * 16 + ln;

  v8f acc[5] = {};
  const float* arow = U3 + row * K3C;
  const unsigned short* bfrag0 = w3f + lane * 16;

  // kt = 0..38 : all K <= 1247 < 1270, fully unguarded, vector loads
  for (int kt = 0; kt < 39; ++kt) {
    const float2* lo = (const float2*)(arow + kt * 32 + 8 * h); // 8B aligned
    v16bf a;
    #pragma unroll
    for (int j = 0; j < 4; ++j) {
      float2 x = lo[j];          // K = kb + 2j, +1
      a[2*j]     = (__bf16)x.x;
      a[2*j + 1] = (__bf16)x.y;
      float2 y = lo[8 + j];      // K = kb + 16 + 2j, +1
      a[8 + 2*j]     = (__bf16)y.x;
      a[8 + 2*j + 1] = (__bf16)y.y;
    }
    #pragma unroll
    for (int ct = 0; ct < 5; ++ct) {
      v16bf b = *(const v16bf*)(bfrag0 + (kt * 5 + ct) * FRAG);
      acc[ct] = __builtin_amdgcn_wmma_f32_16x16x32_bf16(
          false, a, false, b, (short)0, acc[ct], false, false);
    }
  }
  // kt = 39 tail: clamp-and-select (no exec-mask branching)
  {
    const int kb = 39 * 32 + 8 * h;
    v16bf a;
    #pragma unroll
    for (int e = 0; e < 16; ++e) {
      int k  = kb + (e < 8 ? e : e + 8);
      int ki = (k < K3C) ? k : 0;
      float x = arow[ki];
      a[e] = (__bf16)((k < K3C) ? x : 0.f);
    }
    #pragma unroll
    for (int ct = 0; ct < 5; ++ct) {
      v16bf b = *(const v16bf*)(bfrag0 + (39 * 5 + ct) * FRAG);
      acc[ct] = __builtin_amdgcn_wmma_f32_16x16x32_bf16(
          false, a, false, b, (short)0, acc[ct], false, false);
    }
  }

  // Scatter D (lane=n=c-col, VGPR r -> M = 8*h + r) into blob as bf16.
  #pragma unroll
  for (int ct = 0; ct < 5; ++ct) {
    const int c = ct * 16 + ln;
    #pragma unroll
    for (int r = 0; r < 8; ++r) {
      int rowd = rt * 16 + 8 * h + r;
      int p  = rowd / PQ;
      int qi = rowd - p * PQ;
      blob[blob_idx(c, p, qi)] = bfbits(acc[ct][r]);
    }
  }
}

// ---------------------------------------------------------------------------
// Kernel 2: fill appended K' columns: [2304..2351] = UW2[c,p,q],
// [2352] = UW1[c,p], [2353..2367] = 0.  Tiny (245k elements, <=24 MACs each).
// ---------------------------------------------------------------------------
__global__ void k_uw21(const float* __restrict__ U2, const float* __restrict__ W2,
                       const float* __restrict__ U1, const float* __restrict__ W1,
                       unsigned short* __restrict__ blob)
{
  int t = blockIdx.x * 256 + threadIdx.x;
  if (t >= NC * 3 * 2 * 512) return;
  int e    = t & 15;
  int lane = (t >> 4) & 31;
  int kt2  = (t >> 9) & 1;
  int mtc  = t >> 10;
  int mt   = mtc % 3;
  int c    = mtc / 3;
  int h = lane >> 4, ln = lane & 15;
  int p = mt * 16 + ln;
  int K = (72 + kt2) * 32 + kmap(e, h);
  int q = K - PQ;                          // 0..63
  float v = 0.f;
  if (q < 48) {
    const float* u2 = U2 + (p * 48 + q) * 24;
    float s = 0.f;
    #pragma unroll
    for (int k = 0; k < 24; ++k) s += u2[k] * W2[k * NC + c];
    v = s;
  } else if (q == 48) {
    float s = 0.f;
    #pragma unroll
    for (int k = 0; k < 3; ++k) s += U1[p * 3 + k] * W1[k * NC + c];
    v = s;
  }
  blob[((long)((c * 3 + mt) * NKT + 72 + kt2) * 32 + lane) * 16 + e] = bfbits(v);
}

// ---------------------------------------------------------------------------
// Kernel 3: main contraction. One wave per (c, 16-atom tile).
//  - stage f[16][48] to LDS
//  - build bf16 outer-product tile op[16][2368] in LDS (B operand source;
//    exploits CDNA5's 320 KB LDS, fragments load as two aligned ds_load_b128)
//  - 74 k-tiles x 3 p-tiles of v_wmma_f32_16x16x32_bf16
//  - epilogue: sum_p f_p * t[p,n] via ds_swizzle SWAPX16 cross-half reduce
// ---------------------------------------------------------------------------
__global__ void __launch_bounds__(32)
k_contract(const float* __restrict__ nf, const unsigned short* __restrict__ blob,
           float* __restrict__ out)
{
  extern __shared__ char smem[];
  unsigned short* op  = (unsigned short*)smem;           // [16][KP] bf16
  float*          fsh = (float*)(smem + 16 * KP * 2);    // [16][48] f32

  const int lane = threadIdx.x;
  const int c    = blockIdx.x;        // 0..79
  const int n0   = blockIdx.y * 16;   // atom tile base
  const int h    = lane >> 4;
  const int ln   = lane & 15;

  if (lane < 16) {
    const float4* src = (const float4*)(nf + ((long)(n0 + lane) * NC + c) * NI);
    float4* dst = (float4*)(fsh + lane * NI);
    #pragma unroll
    for (int j = 0; j < 12; ++j) dst[j] = src[j];
  }
  __syncthreads();

  // f of "my" atom in registers (static indices for the unrolled i-loop)
  float fr[48];
  {
    const float4* s = (const float4*)(fsh + ln * NI);
    #pragma unroll
    for (int j = 0; j < 12; ++j) {
      float4 v = s[j];
      fr[4*j] = v.x; fr[4*j+1] = v.y; fr[4*j+2] = v.z; fr[4*j+3] = v.w;
    }
  }

  // Build outer-product rows: lanes h=0 do q=0..23, h=1 do q=24..47 + tail.
  unsigned short* oprow = op + ln * KP;
  const int q0 = h * 24;
  for (int q = q0; q < q0 + 24; ++q) {
    float fq = fsh[ln * NI + q];
    unsigned tmp[24];
    #pragma unroll
    for (int i = 0; i < 24; ++i) tmp[i] = pack2(fq * fr[2*i], fq * fr[2*i+1]);
    uint4* d = (uint4*)(oprow + q * 48);
    #pragma unroll
    for (int j = 0; j < 6; ++j)
      d[j] = make_uint4(tmp[4*j], tmp[4*j+1], tmp[4*j+2], tmp[4*j+3]);
  }
  if (h) {  // K' 2304..2335 : f_q, q = 0..31 (2336..2367 unwritten)
    unsigned tmp[16];
    #pragma unroll
    for (int j = 0; j < 16; ++j) {
      float a = (2*j < 48) ? fr[2*j] : (2*j == 48 ? 1.f : 0.f);
      float b = (2*j + 1 < 48) ? fr[2*j+1] : 0.f;
      tmp[j] = pack2(a, b);
    }
    uint4* d = (uint4*)(oprow + PQ);
    #pragma unroll
    for (int j = 0; j < 4; ++j)
      d[j] = make_uint4(tmp[4*j], tmp[4*j+1], tmp[4*j+2], tmp[4*j+3]);
  }
  __syncthreads();

  v8f acc0 = {}, acc1 = {}, acc2 = {};
  const unsigned short* A0 = blob + ((long)c * 3) * NKT * FRAG + lane * 16;
  const unsigned short* bbase = op + ln * KP + 8 * h;

  #pragma unroll 2
  for (int kt = 0; kt < NKT; ++kt) {
    v16bf a0 = *(const v16bf*)(A0 + (long)kt * FRAG);
    v16bf a1 = *(const v16bf*)(A0 + ((long)NKT + kt) * FRAG);
    v16bf a2 = *(const v16bf*)(A0 + ((long)2 * NKT + kt) * FRAG);
    const unsigned short* bs = bbase + kt * 32;
    uint4 blo = *(const uint4*)bs;            // K = kt*32+8h .. +8
    uint4 bhi = *(const uint4*)(bs + 16);     // K = kt*32+8h+16 .. +24
    v8u bu;
    bu[0] = blo.x; bu[1] = blo.y; bu[2] = blo.z; bu[3] = blo.w;
    bu[4] = bhi.x; bu[5] = bhi.y; bu[6] = bhi.z; bu[7] = bhi.w;
    v16bf b = __builtin_bit_cast(v16bf, bu);
    acc0 = __builtin_amdgcn_wmma_f32_16x16x32_bf16(false, a0, false, b, (short)0, acc0, false, false);
    acc1 = __builtin_amdgcn_wmma_f32_16x16x32_bf16(false, a1, false, b, (short)0, acc1, false, false);
    acc2 = __builtin_amdgcn_wmma_f32_16x16x32_bf16(false, a2, false, b, (short)0, acc2, false, false);
  }

  // out[n,c] = sum_p f_p * t[p,n] ; lane holds p = mt*16 + 8h + r for atom ln.
  float partial = 0.f;
  #pragma unroll
  for (int r = 0; r < 8; ++r) {
    int pb = 8 * h + r;
    partial += acc0[r] * fsh[ln * NI + pb]
             + acc1[r] * fsh[ln * NI + 16 + pb]
             + acc2[r] * fsh[ln * NI + 32 + pb];
  }
  int swz = __builtin_amdgcn_ds_swizzle(__float_as_int(partial), 0x401F); // SWAPX16
  float total = partial + __int_as_float(swz);
  if (lane < 16) out[(long)(n0 + ln) * NC + c] = total;
}

// ---------------------------------------------------------------------------
extern "C" void kernel_launch(void* const* d_in, const int* in_sizes, int n_in,
                              void* d_out, int out_size, void* d_ws, size_t ws_size,
                              hipStream_t stream)
{
  const float* nf = (const float*)d_in[0];
  const float* U3 = (const float*)d_in[1];
  const float* U2 = (const float*)d_in[2];
  const float* U1 = (const float*)d_in[3];
  const float* W3 = (const float*)d_in[4];
  const float* W2 = (const float*)d_in[5];
  const float* W1 = (const float*)d_in[6];
  float* out = (float*)d_out;
  unsigned short* blob = (unsigned short*)d_ws;            // 18,186,240 B
  unsigned short* w3f  = blob + BLOB_ELEMS;                // + 204,800 B

  (void)in_sizes; (void)n_in; (void)out_size; (void)ws_size;

  k_w3frag<<<dim3((W3F_ELEMS + 255) / 256), 256, 0, stream>>>(W3, w3f);
  k_uw21<<<dim3((NC * 3 * 2 * 512 + 255) / 256), 256, 0, stream>>>(U2, W2, U1, W1, blob);
  k_uw3<<<dim3((48 * PQ) / 16), 32, 0, stream>>>(U3, w3f, blob);

  size_t shmem = (size_t)16 * KP * 2 + (size_t)16 * NI * 4;  // 78,848 B
  k_contract<<<dim3(NC, NA / 16), 32, shmem, stream>>>(nf, blob, out);
}
